// DGMoE_15315853378164
// MI455X (gfx1250) — compile-verified, matrix operand order checked by reading, not run on hardware
//
#include <hip/hip_runtime.h>
#include <hip/hip_bf16.h>
#include <math.h>

// ---------------------------------------------------------------------------
// DGMoE for gfx1250 (MI455X): bf16 WMMA (16x16x32) + Tensor Data Mover staging.
//   out = ( y[8192,1024], logits[8192,8] )  fp32, concatenated.
//
// Pipeline:
//   1. cvt   : Xbf = bf16(x)
//   2. tcvt  : W1t[e] = bf16(W1[e]^T) [F,D] ; W2t[e] = bf16(W2[e]^T) [D,F]
//   3. router: logits, gg[t,e] = g/(sum g + eps)
//   4. per expert: H = gelu(Xbf @ W1t^T + b1)  ;  y (+)= gg * (H @ W2t^T + b2)
//
// GEMM staging: TENSOR_LOAD_TO_LDS (TDM) with hardware LDS padding and a
// statically-indexed LDS ping-pong (K loop unrolled by 2 slabs) so fragment
// loads keep literal ds offsets.  s_wait_tensorcnt pipelines one slab ahead.
// Falls back to register-staged copies if the TDM builtin is unavailable.
// ---------------------------------------------------------------------------

typedef __attribute__((ext_vector_type(16))) __bf16 v16bf;
typedef __attribute__((ext_vector_type(8)))  float  v8f;
typedef __attribute__((ext_vector_type(8)))  __bf16 bf16x8;
typedef __attribute__((ext_vector_type(2)))  __bf16 bf16x2;
typedef __attribute__((ext_vector_type(4)))  float  f32x4;
typedef __attribute__((ext_vector_type(4)))  unsigned int u32x4;
typedef __attribute__((ext_vector_type(8)))  int    i32x8;
typedef __attribute__((ext_vector_type(4)))  int    i32x4;

constexpr int T = 4 * 2048;   // tokens
constexpr int D = 1024;
constexpr int E = 8;
constexpr int F = 4096;

constexpr int BM = 128;       // token tile
constexpr int BN = 128;       // output-feature tile
constexpr int BK = 64;        // staged K slab (2 WMMA K-steps)
constexpr int LDK = BK + 8;   // padded LDS row stride (bank-conflict avoidance)

#if defined(__AMDGCN__) && __has_builtin(__builtin_amdgcn_tensor_load_to_lds) && \
    __has_builtin(__builtin_amdgcn_s_wait_tensorcnt)
#define HAVE_TDM 1
#else
#define HAVE_TDM 0
#endif

#if HAVE_TDM
#if __has_include(<hip/amd_detail/amd_gfx1250_TDM.h>)
// therock/clang-23 toolchain: 6-arg form (extra int32x8 group)
#define TDM_ISSUE(g0, g1, g2, g3)                                         \
  __builtin_amdgcn_tensor_load_to_lds((g0), (g1), (g2), (g3),             \
      (i32x8){0, 0, 0, 0, 0, 0, 0, 0}, 0)
#else
// ROCm 7.2 / clang-22: 5-arg form
#define TDM_ISSUE(g0, g1, g2, g3) \
  __builtin_amdgcn_tensor_load_to_lds((g0), (g1), (g2), (g3), 0)
#endif

// LDS byte offset of a __shared__ object: per CDNA5 aperture mapping the
// hardware LDS address is addr[31:0] of the generic pointer.
__device__ __forceinline__ unsigned lds_byte_off(const void* p) {
  return (unsigned)(unsigned long long)(uintptr_t)p;
}

// Issue one TDM 2D tile load: tile_rows x BK bf16 (k-contiguous) from a
// row-major bf16 tensor with row stride `stride` (elements) into LDS with
// hardware row padding 64 data + 8 pad bf16 (pad_interval=32 DW, pad=4 DW),
// reproducing the LDK=72 layout.  D# bit layout per CDNA5 ISA 8.3/8.4.
__device__ __forceinline__ void tdm_load_tile(const __bf16* gsrc,
                                              unsigned lds_off,
                                              unsigned stride,
                                              unsigned rows_total,
                                              unsigned tile_rows) {
  unsigned long long ga = (unsigned long long)(uintptr_t)gsrc;
  u32x4 g0;
  g0[0] = 1u;                                                // count=1 (valid D#)
  g0[1] = lds_off;                                           // lds_addr (bytes)
  g0[2] = (unsigned)(ga & 0xFFFFFFFFu);                      // global_addr lo
  g0[3] = (unsigned)((ga >> 32) & 0x01FFFFFFu) | (2u << 30); // addr hi | type=2
  i32x8 g1;
  g1[0] = (1 << 16)      // data_size = 2 bytes (bf16)
        | (1 << 20)      // pad_enable
        | (4 << 22)      // pad_interval: 2^(4+1)=32 DWORDs (=64 bf16 = one row)
        | (3 << 25);     // pad_amount: 4 DWORDs (=8 bf16)  -> LDS stride 72
  g1[1] = (int)((stride & 0xFFFFu) << 16);                   // tensor_dim0[15:0]
  g1[2] = (int)(((stride >> 16) & 0xFFFFu) |
                ((rows_total & 0xFFFFu) << 16));             // dim0 hi | dim1 lo
  g1[3] = (int)(((rows_total >> 16) & 0xFFFFu) |
                ((unsigned)BK << 16));                       // dim1 hi | tile_dim0
  g1[4] = (int)(tile_rows & 0xFFFFu);                        // tile_dim1 (dim2=0)
  g1[5] = (int)(stride);                                     // dim0_stride lo
  g1[6] = 0;                                                 // stride hi | dim1_stride
  g1[7] = 0;
  i32x4 z = {0, 0, 0, 0};
  TDM_ISSUE(g0, g1, z, z);
}
#endif  // HAVE_TDM

// ---------------------------------------------------------------------------
// 16x32 bf16 WMMA operand fragment from an LDS tile (K contiguous).
// CDNA5 ISA 7.12.2 lane map; pairs merge into ds_load_b128.
// ---------------------------------------------------------------------------
__device__ __forceinline__ v16bf load_frag16(const __bf16* rowptr, int half) {
  v16bf f;
#pragma unroll
  for (int v = 0; v < 8; ++v) {
    int kk = ((v & 3) << 1) + ((v >> 2) << 4) + (half << 3);
    bf16x2 p = *(const bf16x2*)(rowptr + kk);
    f[2 * v]     = p[0];
    f[2 * v + 1] = p[1];
  }
  return f;
}

// One BK=64 slab of WMMA work for this wave's 64x32 patch (4x2 C tiles).
__device__ __forceinline__ void compute_slab(const __bf16 (*As)[LDK],
                                             const __bf16 (*Bs)[LDK],
                                             int wrow, int wcol, int l16,
                                             int half, v8f acc[4][2]) {
#pragma unroll
  for (int kk = 0; kk < BK; kk += 32) {
    v16bf afr[4], bfr[2];
#pragma unroll
    for (int tm = 0; tm < 4; ++tm)
      afr[tm] = load_frag16(&As[wrow * 64 + tm * 16 + l16][kk], half);
#pragma unroll
    for (int tn = 0; tn < 2; ++tn)
      bfr[tn] = load_frag16(&Bs[wcol * 32 + tn * 16 + l16][kk], half);
#pragma unroll
    for (int tm = 0; tm < 4; ++tm)
#pragma unroll
      for (int tn = 0; tn < 2; ++tn)
        acc[tm][tn] = __builtin_amdgcn_wmma_f32_16x16x32_bf16(
            false, afr[tm], false, bfr[tn], (short)0, acc[tm][tn], false, false);
  }
}

constexpr int NBUF = HAVE_TDM ? 4 : 2;   // [buf][A/B] tiles (TDM ping-pong)

// ---------------------------------------------------------------------------
// 128x128 block GEMM core, A[m][k] x B[n][k]^T, bf16 in / f32 acc.
// 8 waves; wave (wrow,wcol) owns a 64x32 patch = 4x2 C tiles.
// ---------------------------------------------------------------------------
__device__ __forceinline__ void gemm_tile_bf16(
    const __bf16* __restrict__ Ag, int lda, int arows,
    const __bf16* __restrict__ Bg, int ldb, int brows,
    int Kdim, int m0, int n0,
    __bf16 (*smem)[BM][LDK],
    v8f acc[4][2]) {
  const int tid  = threadIdx.x;
  const int lane = tid & 31;
  const int wave = tid >> 5;
  const int wrow = wave >> 2;       // 0..1
  const int wcol = wave & 3;        // 0..3
  const int l16  = lane & 15;
  const int half = lane >> 4;

#if HAVE_TDM
  // ---- Tensor Data Mover staging: wave 0 drives the DMA engine.
  // Statically indexed ping-pong (loop unrolled by 2 slabs) keeps LDS
  // fragment addresses as literal ds offsets.
  const __bf16* aP = Ag + (size_t)m0 * lda;
  const __bf16* bP = Bg + (size_t)n0 * ldb;
  const bool issuer = (wave == 0);
  const int  nslab  = Kdim / BK;    // even (16 or 64)

  if (issuer) {  // prologue: slab 0 -> buf0, slab 1 -> buf1
    tdm_load_tile(aP,      lds_byte_off(&smem[0][0][0]), (unsigned)lda,
                  (unsigned)arows, BM);
    tdm_load_tile(bP,      lds_byte_off(&smem[1][0][0]), (unsigned)ldb,
                  (unsigned)brows, BN);
    tdm_load_tile(aP + BK, lds_byte_off(&smem[2][0][0]), (unsigned)lda,
                  (unsigned)arows, BM);
    tdm_load_tile(bP + BK, lds_byte_off(&smem[3][0][0]), (unsigned)ldb,
                  (unsigned)brows, BN);
  }

  for (int i = 0; i < nslab; i += 2) {
    const int kA = i * BK;          // slab i  (buf0)
    const int kB = kA + BK;         // slab i+1 (buf1)

    // ---- phase A: compute slab i from buf0 -----------------------------
    if (issuer) __builtin_amdgcn_s_wait_tensorcnt(2);   // slab i landed
    __syncthreads();
    compute_slab(smem[0], smem[1], wrow, wcol, l16, half, acc);
    __syncthreads();
    if (issuer) {
      if (i + 2 < nslab) {          // refill buf0 with slab i+2
        tdm_load_tile(aP + kA + 2 * BK, lds_byte_off(&smem[0][0][0]),
                      (unsigned)lda, (unsigned)arows, BM);
        tdm_load_tile(bP + kA + 2 * BK, lds_byte_off(&smem[1][0][0]),
                      (unsigned)ldb, (unsigned)brows, BN);
        __builtin_amdgcn_s_wait_tensorcnt(2);           // slab i+1 landed
      } else {
        __builtin_amdgcn_s_wait_tensorcnt(0);           // last pair: drain
      }
    }

    // ---- phase B: compute slab i+1 from buf1 ---------------------------
    __syncthreads();
    compute_slab(smem[2], smem[3], wrow, wcol, l16, half, acc);
    __syncthreads();
    if (issuer && (i + 3 < nslab)) {                    // refill buf1
      tdm_load_tile(aP + kB + 2 * BK, lds_byte_off(&smem[2][0][0]),
                    (unsigned)lda, (unsigned)arows, BM);
      tdm_load_tile(bP + kB + 2 * BK, lds_byte_off(&smem[3][0][0]),
                    (unsigned)ldb, (unsigned)brows, BN);
    }
  }
#else
  // ---- Fallback: register-double-buffered global->LDS staging ------------
  __bf16(*As)[LDK] = smem[0];
  __bf16(*Bs)[LDK] = smem[1];
  const int row = tid >> 1;
  const int ks  = (tid & 1) << 5;

  const __bf16* aG = Ag + (size_t)(m0 + row) * lda + ks;
  const __bf16* bG = Bg + (size_t)(n0 + row) * ldb + ks;

  bf16x8 sa[4], sb[4];
#pragma unroll
  for (int i = 0; i < 4; ++i) {
    sa[i] = *(const bf16x8*)(aG + 8 * i);
    sb[i] = *(const bf16x8*)(bG + 8 * i);
  }

  for (int k0 = 0; k0 < Kdim; k0 += BK) {
#pragma unroll
    for (int i = 0; i < 4; ++i) {
      *(bf16x8*)(&As[row][ks] + 8 * i) = sa[i];
      *(bf16x8*)(&Bs[row][ks] + 8 * i) = sb[i];
    }
    __syncthreads();
    if (k0 + BK < Kdim) {
#pragma unroll
      for (int i = 0; i < 4; ++i) {
        sa[i] = *(const bf16x8*)(aG + (k0 + BK) + 8 * i);
        sb[i] = *(const bf16x8*)(bG + (k0 + BK) + 8 * i);
      }
    }
    compute_slab(As, Bs, wrow, wcol, l16, half, acc);
    __syncthreads();
  }
#endif
}

// ---------------------------------------------------------------------------
// Prep: elementwise fp32 -> bf16 (8 elems/thread, vectorized).
// ---------------------------------------------------------------------------
__global__ __launch_bounds__(256)
void cvt_f32_to_bf16(const float* __restrict__ in, __bf16* __restrict__ out) {
  size_t i = ((size_t)blockIdx.x * 256 + threadIdx.x) * 8;
  f32x4 a = *(const f32x4*)(in + i);
  f32x4 b = *(const f32x4*)(in + i + 4);
  bf16x8 o;
#pragma unroll
  for (int j = 0; j < 4; ++j) { o[j] = (__bf16)a[j]; o[4 + j] = (__bf16)b[j]; }
  *(bf16x8*)(out + i) = o;
}

// ---------------------------------------------------------------------------
// Prep: per-expert transpose + convert. in: [R][C] fp32, out: [C][R] bf16.
// ---------------------------------------------------------------------------
__global__ __launch_bounds__(256)
void tcvt_f32_to_bf16(const float* __restrict__ in, __bf16* __restrict__ out,
                      int R, int C) {
  __shared__ float t[32][33];
  const float* inp  = in  + (size_t)blockIdx.z * R * C;
  __bf16*      outp = out + (size_t)blockIdx.z * R * C;
  const int tx = threadIdx.x & 31, ty = threadIdx.x >> 5;
  const int r0 = blockIdx.y * 32, c0 = blockIdx.x * 32;
#pragma unroll
  for (int i = 0; i < 4; ++i) {
    int r = ty + i * 8;
    t[r][tx] = inp[(size_t)(r0 + r) * C + (c0 + tx)];
  }
  __syncthreads();
#pragma unroll
  for (int i = 0; i < 4; ++i) {
    int c = ty + i * 8;
    outp[(size_t)(c0 + c) * R + (r0 + tx)] = (__bf16)t[tx][c];
  }
}

// ---------------------------------------------------------------------------
// Router: one wave per token (wave32), fp32.
// ---------------------------------------------------------------------------
__global__ __launch_bounds__(256)
void dgmoe_router(const float* __restrict__ x, const float* __restrict__ prev,
                  const float* __restrict__ Wt, const float* __restrict__ Wgt,
                  const float* __restrict__ Wel,
                  float* __restrict__ logits_out, float* __restrict__ gg) {
  const int lane  = threadIdx.x & 31;
  const int token = blockIdx.x * 8 + (threadIdx.x >> 5);

  float acc[E];
#pragma unroll
  for (int e = 0; e < E; ++e) acc[e] = 0.f;

  const float* xrow = x + (size_t)token * D;
  for (int d = lane; d < D; d += 32) {
    float xv = xrow[d];
#pragma unroll
    for (int e = 0; e < E; ++e) acc[e] += xv * Wt[e * D + d];
  }
#pragma unroll
  for (int off = 16; off > 0; off >>= 1)
#pragma unroll
    for (int e = 0; e < E; ++e) acc[e] += __shfl_xor(acc[e], off, 32);

  if (lane == 0) {
    float lg[E];
#pragma unroll
    for (int e = 0; e < E; ++e) {
      float s = acc[e];
#pragma unroll
      for (int j = 0; j < E; ++j) s += prev[(size_t)token * E + j] * Wgt[e * E + j];
      lg[e] = s;
    }
    float mx = lg[0];
#pragma unroll
    for (int e = 1; e < E; ++e) mx = fmaxf(mx, lg[e]);
    float st[E], ssum = 0.f;
#pragma unroll
    for (int e = 0; e < E; ++e) { st[e] = expf(lg[e] - mx); ssum += st[e]; }
    float inv_s = 1.f / ssum, den = 0.f, g[E];
#pragma unroll
    for (int e = 0; e < E; ++e) {
      float p  = st[e] * inv_s;
      float We = 1.f / (1.f + expf(-Wel[e]));
      float gv = (p > 0.5f * We) ? p : 0.f;
      g[e] = gv;
      den += gv;
    }
    float inv_d = 1.f / (den + 1e-6f);
#pragma unroll
    for (int e = 0; e < E; ++e) {
      logits_out[(size_t)token * E + e] = lg[e];
      gg[(size_t)token * E + e]         = g[e] * inv_d;
    }
  }
}

// ---------------------------------------------------------------------------
// GEMM1 + exact GELU: H = gelu(Xbf @ W1t[e]^T + b1[e])  (bf16)
// ---------------------------------------------------------------------------
__global__ __launch_bounds__(256)
void dgmoe_gemm1_gelu(const __bf16* __restrict__ Xbf,
                      const __bf16* __restrict__ W1t,
                      const float* __restrict__ b1,
                      __bf16* __restrict__ H, int e) {
  __shared__ __bf16 smem[NBUF][BM][LDK];

  const int m0 = blockIdx.x * BM;
  const int n0 = blockIdx.y * BN;

  v8f acc[4][2] = {};
  gemm_tile_bf16(Xbf, D, T, W1t + (size_t)e * F * D, D, F, D, m0, n0, smem, acc);

  const int lane = threadIdx.x & 31;
  const int wave = threadIdx.x >> 5;
  const int wrow = wave >> 2, wcol = wave & 3;
  const int l16  = lane & 15, half = lane >> 4;

#pragma unroll
  for (int tn = 0; tn < 2; ++tn) {
    const int   n  = n0 + wcol * 32 + tn * 16 + l16;
    const float bv = b1[(size_t)e * F + n];
#pragma unroll
    for (int tm = 0; tm < 4; ++tm) {
#pragma unroll
      for (int r = 0; r < 8; ++r) {
        int   m   = m0 + wrow * 64 + tm * 16 + r + half * 8;
        float v   = acc[tm][tn][r] + bv;
        float gel = 0.5f * v * (1.0f + erff(v * 0.70710678118654752440f));
        H[(size_t)m * F + n] = (__bf16)gel;
      }
    }
  }
}

// ---------------------------------------------------------------------------
// GEMM2 + gated accumulate: y (+)= gg[:,e] * (H @ W2t[e]^T + b2[e])
// ---------------------------------------------------------------------------
__global__ __launch_bounds__(256)
void dgmoe_gemm2_acc(const __bf16* __restrict__ H,
                     const __bf16* __restrict__ W2t,
                     const float* __restrict__ b2,
                     const float* __restrict__ gg,
                     float* __restrict__ y, int e, int first) {
  __shared__ __bf16 smem[NBUF][BM][LDK];

  const int m0 = blockIdx.x * BM;
  const int n0 = blockIdx.y * BN;

  v8f acc[4][2] = {};
  gemm_tile_bf16(H, F, T, W2t + (size_t)e * F * D, F, D, F, m0, n0, smem, acc);

  const int lane = threadIdx.x & 31;
  const int wave = threadIdx.x >> 5;
  const int wrow = wave >> 2, wcol = wave & 3;
  const int l16  = lane & 15, half = lane >> 4;

#pragma unroll
  for (int tn = 0; tn < 2; ++tn) {
    const int   n  = n0 + wcol * 32 + tn * 16 + l16;
    const float bv = b2[(size_t)e * D + n];
#pragma unroll
    for (int tm = 0; tm < 4; ++tm) {
#pragma unroll
      for (int r = 0; r < 8; ++r) {
        int    m   = m0 + wrow * 64 + tm * 16 + r + half * 8;
        float  ggv = gg[(size_t)m * E + e];
        size_t idx = (size_t)m * D + n;
        float  v   = ggv * (acc[tm][tn][r] + bv);
        float  pv  = first ? 0.f : y[idx];
        y[idx] = pv + v;
      }
    }
  }
}

// ---------------------------------------------------------------------------
extern "C" void kernel_launch(void* const* d_in, const int* in_sizes, int n_in,
                              void* d_out, int out_size, void* d_ws, size_t ws_size,
                              hipStream_t stream) {
  const float* x    = (const float*)d_in[0];
  const float* prev = (const float*)d_in[1];
  const float* Wt   = (const float*)d_in[2];
  const float* Wgt  = (const float*)d_in[3];
  const float* Wel  = (const float*)d_in[4];
  const float* W1   = (const float*)d_in[5];
  const float* b1   = (const float*)d_in[6];
  const float* W2   = (const float*)d_in[7];
  const float* b2   = (const float*)d_in[8];

  float* y      = (float*)d_out;          // [T, D]
  float* logits = y + (size_t)T * D;      // [T, E]

  char*   w    = (char*)d_ws;
  float*  gg   = (float*)w;               w += (size_t)T * E * sizeof(float);
  __bf16* Xbf  = (__bf16*)w;              w += (size_t)T * D * sizeof(__bf16);
  __bf16* W1t  = (__bf16*)w;              w += (size_t)E * D * F * sizeof(__bf16);
  __bf16* W2t  = (__bf16*)w;              w += (size_t)E * F * D * sizeof(__bf16);
  __bf16* H    = (__bf16*)w;

  cvt_f32_to_bf16<<<(T * D) / (256 * 8), 256, 0, stream>>>(x, Xbf);
  tcvt_f32_to_bf16<<<dim3(F / 32, D / 32, E), 256, 0, stream>>>(W1, W1t, D, F);
  tcvt_f32_to_bf16<<<dim3(D / 32, F / 32, E), 256, 0, stream>>>(W2, W2t, F, D);

  dgmoe_router<<<T / 8, 256, 0, stream>>>(x, prev, Wt, Wgt, Wel, logits, gg);

  dim3 g1(T / BM, F / BN);   // 64 x 32
  dim3 g2(T / BM, D / BN);   // 64 x 8
  for (int e = 0; e < E; ++e) {
    dgmoe_gemm1_gelu<<<g1, 256, 0, stream>>>(Xbf, W1t, b1, H, e);
    dgmoe_gemm2_acc<<<g2, 256, 0, stream>>>(H, W2t, b2, gg, y, e, (e == 0) ? 1 : 0);
  }
}
